// TripletFaceAttention_89206470738429
// MI455X (gfx1250) — matrix operand order repaired and checked by reading, not compiled, not run on hardware
//
#include <hip/hip_runtime.h>
#include <hip/hip_bf16.h>
#include <math.h>

typedef __attribute__((ext_vector_type(16))) _Float16 v16h;
typedef __attribute__((ext_vector_type(8)))  _Float16 v8h;
typedef __attribute__((ext_vector_type(8)))  float    v8f;

#define DEV static __device__ __forceinline__

constexpr int Bb = 128, Cc = 768, Nn = 256, NH = 12, HD = 64;

DEV int lane_id() { return (int)(threadIdx.x & 31); }

// ---- WMMA helpers (CDNA5 wave32 layouts) ----------------------------------
// A-frag (16x32 f16): lane l holds row m=l&15; k = (l>=16?16:0)+e,
// e0..15 contiguous. Two adjacent 16B chunks.
DEV v16h load_a_frag(const _Float16* base, int ldk) {
  int l = lane_id();
  const _Float16* p = base + (l & 15) * ldk + (l >> 4) * 16;
  v8h lo = *(const v8h*)(p);
  v8h hi = *(const v8h*)(p + 8);
  return __builtin_shufflevector(lo, hi, 0,1,2,3,4,5,6,7,8,9,10,11,12,13,14,15);
}

// B-frag (32x16 f16), memory layout [n][k] with k contiguous:
// lane l holds col n=l&15, k = (l>=16?16:0)+e  -> 32 contiguous bytes.
DEV v16h load_b_frag_kcontig(const _Float16* base, int ldn) {
  int l = lane_id();
  return *(const v16h*)(base + (l & 15) * ldn + (l >> 4) * 16);
}

DEV v8f wmma_f16(v16h a, v16h b, v8f c) {
  return __builtin_amdgcn_wmma_f32_16x16x32_f16(false, a, false, b, (short)0, c,
                                                false, false);
}

// 16-lane (half-wave) reductions; matches C-frag row layout (row spans 16 lanes)
DEV float rowmax16(float v) {
  v = fmaxf(v, __shfl_xor(v, 1, 16));
  v = fmaxf(v, __shfl_xor(v, 2, 16));
  v = fmaxf(v, __shfl_xor(v, 4, 16));
  v = fmaxf(v, __shfl_xor(v, 8, 16));
  return v;
}
DEV float rowsum16(float v) {
  v += __shfl_xor(v, 1, 16);
  v += __shfl_xor(v, 2, 16);
  v += __shfl_xor(v, 4, 16);
  v += __shfl_xor(v, 8, 16);
  return v;
}

// ---- Prep kernels ----------------------------------------------------------
__global__ __launch_bounds__(256) void cvt_f32_f16(const float* __restrict__ s,
                                                   _Float16* __restrict__ d, int n) {
  int i = blockIdx.x * 256 + threadIdx.x;
  if (i < n) d[i] = (_Float16)s[i];
}

// combined_bias[h][n][m] = face_prior[0][h][n][m] + rel_table[idx(n,m)][h]
__global__ __launch_bounds__(256) void bias_combine(const float* __restrict__ fp,
                                                    const float* __restrict__ rel,
                                                    float* __restrict__ bias) {
  int h = blockIdx.x >> 8;          // 12
  int n = blockIdx.x & 255;         // 256
  int m = threadIdx.x;              // 256
  int y1 = n >> 4, x1 = n & 15, y2 = m >> 4, x2 = m & 15;
  int idx = (y1 - y2 + 15) * 31 + (x1 - x2 + 15);
  size_t o = ((size_t)h * Nn + n) * Nn + m;
  bias[o] = fp[o] + rel[idx * NH + h];
}

// x [B][C][N] f32 -> xt [B][N][C] f16 (tiled transpose)
__global__ __launch_bounds__(256) void transpose_x(const float* __restrict__ x,
                                                   _Float16* __restrict__ xt) {
  __shared__ float tile[32][33];
  int b = blockIdx.x, c0 = blockIdx.y * 32, n0 = blockIdx.z * 32;
  int tx = threadIdx.x & 31, ty = threadIdx.x >> 5;  // 32 x 8
#pragma unroll
  for (int i = 0; i < 4; ++i) {
    int c = c0 + ty + 8 * i;
    tile[ty + 8 * i][tx] = x[((size_t)b * Cc + c) * Nn + n0 + tx];
  }
  __syncthreads();
#pragma unroll
  for (int i = 0; i < 4; ++i) {
    int n = n0 + ty + 8 * i;
    xt[((size_t)b * Nn + n) * Cc + c0 + tx] = (_Float16)tile[tx][ty + 8 * i];
  }
}

// ---- QKV projection: xt[32768x768] @ wq16^T[768x2304] + b -----------------
// grid (36, 256), 256 thr (8 waves); block tile 128m x 64n; wave tile 32x32.
__global__ __launch_bounds__(256) void qkv_gemm(const _Float16* __restrict__ xt,
                                                const _Float16* __restrict__ w16,
                                                const float* __restrict__ bvec,
                                                _Float16* __restrict__ Q,
                                                _Float16* __restrict__ K,
                                                _Float16* __restrict__ Vt) {
  int nb = blockIdx.x, mb = blockIdx.y;
  int wave = threadIdx.x >> 5, wm = wave >> 1, wn = wave & 1;
  int l = lane_id(), lhalf = l >> 4, lcol = l & 15;
  int m0 = mb * 128 + wm * 32;
  int n0 = nb * 64 + wn * 32;

  v8f acc[2][2] = {};
  for (int k = 0; k < Cc; k += 32) {
    v16h af[2], bf[2];
#pragma unroll
    for (int t = 0; t < 2; ++t)
      af[t] = load_a_frag(xt + (size_t)(m0 + t * 16) * Cc + k, Cc);
#pragma unroll
    for (int c = 0; c < 2; ++c)
      bf[c] = load_b_frag_kcontig(w16 + (size_t)(n0 + c * 16) * Cc + k, Cc);
#pragma unroll
    for (int t = 0; t < 2; ++t)
#pragma unroll
      for (int c = 0; c < 2; ++c) acc[t][c] = wmma_f16(af[t], bf[c], acc[t][c]);
  }

  int which = nb / NH, h = nb % NH;
  float bias[2];
#pragma unroll
  for (int c = 0; c < 2; ++c) bias[c] = bvec[n0 + c * 16 + lcol];

#pragma unroll
  for (int t = 0; t < 2; ++t)
#pragma unroll
    for (int c = 0; c < 2; ++c)
#pragma unroll
      for (int r = 0; r < 8; ++r) {
        float v = acc[t][c][r] + bias[c];
        int m = m0 + t * 16 + r + lhalf * 8;   // global row = b*256 + npos
        int bi = m >> 8, np = m & 255;
        int d = wn * 32 + c * 16 + lcol;       // 0..63 within head
        if (which == 0) {
          Q[((size_t)(bi * NH + h) * Nn + np) * HD + d] = (_Float16)(v * 0.125f);
        } else if (which == 1) {
          K[((size_t)(bi * NH + h) * Nn + np) * HD + d] = (_Float16)v;
        } else {
          Vt[((size_t)(bi * NH + h) * HD + d) * Nn + np] = (_Float16)v;
        }
      }
}

// ---- Fused flash attention per (b,h): grid 1536, 8 waves x 32 rows --------
__global__ __launch_bounds__(256) void attn_kernel(const _Float16* __restrict__ Q,
                                                   const _Float16* __restrict__ K,
                                                   const _Float16* __restrict__ Vt,
                                                   const float* __restrict__ bias,
                                                   const float* __restrict__ priors,
                                                   _Float16* __restrict__ Oatt) {
  __shared__ _Float16 plds[8][32][32];  // per-wave P transpose staging (16KB)

  int bh = blockIdx.x;
  int b = bh / NH, h = bh % NH;
  int wave = threadIdx.x >> 5;
  int l = lane_id(), lhalf = l >> 4, lcol = l & 15;
  int rowbase = wave * 32;

  const _Float16* Qb = Q + (size_t)bh * Nn * HD;
  const _Float16* Kb = K + (size_t)bh * Nn * HD;
  const _Float16* Vb = Vt + (size_t)bh * HD * Nn;
  const float* bh_bias = bias + (size_t)h * Nn * Nn;
  const float* b_prior = priors + (size_t)b * Nn * Nn;

  v16h qf[2][2];
#pragma unroll
  for (int t = 0; t < 2; ++t)
#pragma unroll
    for (int kk = 0; kk < 2; ++kk)
      qf[t][kk] = load_a_frag(Qb + (size_t)(rowbase + t * 16) * HD + kk * 32, HD);

  v8f o[2][4] = {};
  float mrow[2][8], lrow[2][8];
#pragma unroll
  for (int t = 0; t < 2; ++t)
#pragma unroll
    for (int r = 0; r < 8; ++r) { mrow[t][r] = -__builtin_inff(); lrow[t][r] = 0.f; }

  for (int kb = 0; kb < Nn; kb += 32) {
    // S = Q K^T  (scale folded into Q)
    v16h kf[2][2];
#pragma unroll
    for (int c = 0; c < 2; ++c)
#pragma unroll
      for (int kk = 0; kk < 2; ++kk)
        kf[c][kk] = load_b_frag_kcontig(Kb + (size_t)(kb + c * 16) * HD + kk * 32, HD);

    v8f s[2][2];
#pragma unroll
    for (int t = 0; t < 2; ++t)
#pragma unroll
      for (int c = 0; c < 2; ++c) {
        v8f z = {};
        z = wmma_f16(qf[t][0], kf[c][0], z);
        s[t][c] = wmma_f16(qf[t][1], kf[c][1], z);
      }

    // bias add, prior multiply (pre-softmax, as reference)
#pragma unroll
    for (int t = 0; t < 2; ++t)
#pragma unroll
      for (int c = 0; c < 2; ++c)
#pragma unroll
        for (int r = 0; r < 8; ++r) {
          int row = rowbase + t * 16 + r + lhalf * 8;
          int col = kb + c * 16 + lcol;
          size_t off = (size_t)row * Nn + col;
          s[t][c][r] = (s[t][c][r] + bh_bias[off]) * b_prior[off];
        }

    // online softmax update
#pragma unroll
    for (int t = 0; t < 2; ++t)
#pragma unroll
      for (int r = 0; r < 8; ++r) {
        float v0 = s[t][0][r], v1 = s[t][1][r];
        float mx = rowmax16(fmaxf(v0, v1));
        float mnew = fmaxf(mrow[t][r], mx);
        float corr = __expf(mrow[t][r] - mnew);
        float p0 = __expf(v0 - mnew), p1 = __expf(v1 - mnew);
        s[t][0][r] = p0; s[t][1][r] = p1;
        lrow[t][r] = lrow[t][r] * corr + rowsum16(p0 + p1);
        mrow[t][r] = mnew;
#pragma unroll
        for (int vc = 0; vc < 4; ++vc) o[t][vc][r] *= corr;
      }

    // P: C-layout -> A-layout via per-wave LDS (DS ops in-order per wave)
#pragma unroll
    for (int t = 0; t < 2; ++t)
#pragma unroll
      for (int c = 0; c < 2; ++c)
#pragma unroll
        for (int r = 0; r < 8; ++r)
          plds[wave][t * 16 + r + lhalf * 8][c * 16 + lcol] = (_Float16)s[t][c][r];

    v16h pf[2];
#pragma unroll
    for (int t = 0; t < 2; ++t) pf[t] = load_a_frag(&plds[wave][t * 16][0], 32);

    // O += P V  (V stored transposed: [d][n], k contiguous)
#pragma unroll
    for (int vc = 0; vc < 4; ++vc) {
      v16h vf = load_b_frag_kcontig(Vb + (size_t)(vc * 16) * Nn + kb, Nn);
#pragma unroll
      for (int t = 0; t < 2; ++t) o[t][vc] = wmma_f16(pf[t], vf, o[t][vc]);
    }
  }

  // epilogue: normalize and write [B][N][C] with c = h*64+d
#pragma unroll
  for (int t = 0; t < 2; ++t)
#pragma unroll
    for (int vc = 0; vc < 4; ++vc)
#pragma unroll
      for (int r = 0; r < 8; ++r) {
        int row = rowbase + t * 16 + r + lhalf * 8;
        int col = h * HD + vc * 16 + lcol;
        Oatt[((size_t)b * Nn + row) * Cc + col] =
            (_Float16)(o[t][vc][r] / lrow[t][r]);
      }
}

// ---- Output projection: Oatt[32768x768] @ wp16^T + b -> out [B][C][N] -----
__global__ __launch_bounds__(256) void proj_gemm(const _Float16* __restrict__ A,
                                                 const _Float16* __restrict__ w16,
                                                 const float* __restrict__ bvec,
                                                 float* __restrict__ out) {
  int nb = blockIdx.x, mb = blockIdx.y;  // (12, 256)
  int wave = threadIdx.x >> 5, wm = wave >> 1, wn = wave & 1;
  int l = lane_id(), lhalf = l >> 4, lcol = l & 15;
  int m0 = mb * 128 + wm * 32;
  int n0 = nb * 64 + wn * 32;

  v8f acc[2][2] = {};
  for (int k = 0; k < Cc; k += 32) {
    v16h af[2], bf[2];
#pragma unroll
    for (int t = 0; t < 2; ++t)
      af[t] = load_a_frag(A + (size_t)(m0 + t * 16) * Cc + k, Cc);
#pragma unroll
    for (int c = 0; c < 2; ++c)
      bf[c] = load_b_frag_kcontig(w16 + (size_t)(n0 + c * 16) * Cc + k, Cc);
#pragma unroll
    for (int t = 0; t < 2; ++t)
#pragma unroll
      for (int c = 0; c < 2; ++c) acc[t][c] = wmma_f16(af[t], bf[c], acc[t][c]);
  }

  float bias[2];
#pragma unroll
  for (int c = 0; c < 2; ++c) bias[c] = bvec[n0 + c * 16 + lcol];

#pragma unroll
  for (int t = 0; t < 2; ++t)
#pragma unroll
    for (int c = 0; c < 2; ++c)
#pragma unroll
      for (int r = 0; r < 8; ++r) {
        int m = m0 + t * 16 + r + lhalf * 8;
        int bi = m >> 8, np = m & 255;
        int cg = n0 + c * 16 + lcol;
        out[((size_t)bi * Cc + cg) * Nn + np] = acc[t][c][r] + bias[c];
      }
}

// ---- Host launch -----------------------------------------------------------
extern "C" void kernel_launch(void* const* d_in, const int* in_sizes, int n_in,
                              void* d_out, int out_size, void* d_ws, size_t ws_size,
                              hipStream_t stream) {
  const float* x           = (const float*)d_in[0];  // [B,768,256]
  const float* face_priors = (const float*)d_in[1];  // [B,256,256]
  const float* qkv_w       = (const float*)d_in[2];  // [2304,768]
  const float* qkv_b       = (const float*)d_in[3];  // [2304]
  const float* rel_table   = (const float*)d_in[4];  // [961,12]
  const float* face_prior  = (const float*)d_in[5];  // [1,12,256,256]
  const float* proj_w      = (const float*)d_in[6];  // [768,768]
  const float* proj_b      = (const float*)d_in[7];  // [768]
  float* out = (float*)d_out;

  char* ws = (char*)d_ws;
  size_t off = 0;
  auto alloc = [&](size_t bytes) {
    char* p = ws + off;
    off += (bytes + 255) & ~(size_t)255;
    return p;
  };
  _Float16* xt   = (_Float16*)alloc((size_t)Bb * Nn * Cc * 2);       // 50 MB
  _Float16* wq16 = (_Float16*)alloc((size_t)3 * Cc * Cc * 2);        // 3.5 MB
  _Float16* wp16 = (_Float16*)alloc((size_t)Cc * Cc * 2);            // 1.2 MB
  float*    cbias= (float*)   alloc((size_t)NH * Nn * Nn * 4);       // 3.1 MB
  _Float16* Q    = (_Float16*)alloc((size_t)Bb * NH * Nn * HD * 2);  // 50 MB
  _Float16* K    = (_Float16*)alloc((size_t)Bb * NH * Nn * HD * 2);  // 50 MB
  _Float16* Vt   = (_Float16*)alloc((size_t)Bb * NH * HD * Nn * 2);  // 50 MB
  _Float16* Oatt = (_Float16*)alloc((size_t)Bb * Nn * Cc * 2);       // 50 MB

  int nqkvw = 3 * Cc * Cc, nprojw = Cc * Cc;
  cvt_f32_f16<<<(nqkvw + 255) / 256, 256, 0, stream>>>(qkv_w, wq16, nqkvw);
  cvt_f32_f16<<<(nprojw + 255) / 256, 256, 0, stream>>>(proj_w, wp16, nprojw);
  bias_combine<<<NH * Nn, 256, 0, stream>>>(face_prior, rel_table, cbias);
  transpose_x<<<dim3(Bb, Cc / 32, Nn / 32), 256, 0, stream>>>(x, xt);

  qkv_gemm<<<dim3(3 * Cc / 64, (Bb * Nn) / 128), 256, 0, stream>>>(
      xt, wq16, qkv_b, Q, K, Vt);

  attn_kernel<<<Bb * NH, 256, 0, stream>>>(Q, K, Vt, cbias, face_priors, Oatt);

  proj_gemm<<<dim3(Cc / 64, (Bb * Nn) / 128), 256, 0, stream>>>(
      Oatt, wp16, proj_b, out);
}
